// DeformConv_1425929142554
// MI455X (gfx1250) — matrix operand order repaired and checked by reading, not compiled, not run on hardware
//
#include <hip/hip_runtime.h>

typedef __attribute__((ext_vector_type(16))) __bf16 v16bf;
typedef __attribute__((ext_vector_type(8)))  float  v8f;

#define HO     64
#define WO     64
#define HW     4096
#define XP     66            // padded spatial extent (1-pixel zero halo)
#define XPC    (XP * XP)     // 4356 padded pixels per image
#define C_TOT  256
#define CG     64
#define KTAPS  9
#define KDIM   576           // deform GEMM K (k-major: col = k*64+i)
#define KOFF   2304          // offset-conv GEMM K (tap-major: K = tap*256 + c)
#define ROWB   584           // KDIM + 8 bf16 pad -> 1168B row stride (bank spread)
#define PTILE  64

__device__ __forceinline__ unsigned short f2bf(float f) {
  unsigned int u = __float_as_uint(f);
  unsigned int r = u + 0x7FFFu + ((u >> 16) & 1u);   // round-to-nearest-even
  return (unsigned short)(r >> 16);
}
__device__ __forceinline__ float bf2f(unsigned int h16) {   // low 16 bits = bf16
  return __uint_as_float(h16 << 16);
}

// ---- prep 1: x (B,C,64,64) f32 -> xtb (B,66,66,C) bf16 with zero halo ----
__global__ void k_transpose(const float* __restrict__ x, unsigned short* __restrict__ xtb) {
  int t = blockIdx.x * blockDim.x + threadIdx.x;        // 4*4356*256 threads
  if (t >= 4 * XPC * C_TOT) return;
  int c = t & 255, s = t >> 8;
  int b = s / XPC, cell = s - b * XPC;
  int yy = cell / XP, xx = cell - yy * XP;
  unsigned short v = 0;
  if (yy >= 1 && yy <= 64 && xx >= 1 && xx <= 64)
    v = f2bf(x[(((size_t)b * C_TOT + c) << 12) + ((yy - 1) << 6) + (xx - 1)]);
  xtb[t] = v;
}

// ---- prep 2a: pack w_deform (C,Cg,3,3) f32 -> A bf16 [c][k*64+i] ----
__global__ void k_pack(const float* __restrict__ w, unsigned short* __restrict__ wp) {
  int t = blockIdx.x * blockDim.x + threadIdx.x;        // 256*576 threads
  int c = t / KDIM, kk = t - c * KDIM;
  int k = kk >> 6, i = kk & 63;
  wp[t] = f2bf(w[(size_t)c * KDIM + i * 9 + k]);
}

// ---- prep 2b: pack w_off (18,256,3,3) f32 -> B bf16 [oc_pad32][tap*256+c] ----
__global__ void k_packoff(const float* __restrict__ woff, unsigned short* __restrict__ wob) {
  int t = blockIdx.x * blockDim.x + threadIdx.x;        // 32*2304 threads
  if (t >= 32 * KOFF) return;
  int oc = t / KOFF, K = t - oc * KOFF;
  int tap = K >> 8, c = K & 255;
  wob[t] = (oc < 18) ? f2bf(woff[((size_t)oc * C_TOT + c) * 9 + tap]) : (unsigned short)0;
}

// ---- offset conv as WMMA GEMM: off[p][oc] = im2col[p][K] x w_off^T[K][oc] ----
// WG = one image row (b, y): M = 64 pixels (4 tiles), N = 32 oc (2 tiles), K = 2304.
// Halo-padded input => no boundary branches in the 72-step hot loop.
__global__ __launch_bounds__(256)
void k_offconv_wmma(const unsigned short* __restrict__ xtb,
                    const unsigned short* __restrict__ wob,
                    const float* __restrict__ boff, float* __restrict__ off) {
  const int row = blockIdx.x;                 // b*64 + y
  const int b = row >> 6, yo = row & 63;
  const int tid = threadIdx.x;
  const int wv = tid >> 5, lane = tid & 31;
  const int lr = lane & 15, half = lane >> 4;
  const int mt = wv & 3, nt = wv >> 2;        // 8 waves -> 4 Mtiles x 2 Ntiles
  const int xo = mt * 16 + lr;                // pixel x for this lane's A rows
  const int kkA = half * 16;                  // A: lane<16 K 0..15; lane>=16 K 16..31
  const int kkB = half * 16;                  // B: lane<16 K 0..15; lane>=16 K 16..31
  union ABU { v16bf v; uint4 q[2]; };
  const unsigned short* brow = wob + (size_t)(nt * 16 + lr) * KOFF;
  const unsigned short* xb   = xtb + (size_t)b * XPC * C_TOT;
  v8f acc = {0, 0, 0, 0, 0, 0, 0, 0};
  #pragma unroll 4
  for (int kt = 0; kt < KOFF / 32; ++kt) {
    const int kb = kt * 32;
    const int tap = kb >> 8;                  // wave-uniform: tile never crosses a tap
    const int di = tap / 3, dj = tap - di * 3;
    const int yp = yo + di, xp = xo + dj;     // padded coords, always in-bounds
    const int c0 = (kb & 255) + kkA;
    ABU a, bb;
    const unsigned short* ap = xb + (size_t)(yp * XP + xp) * C_TOT + c0;
    a.q[0]  = *(const uint4*)(ap);
    a.q[1]  = *(const uint4*)(ap + 8);
    bb.q[0] = *(const uint4*)(brow + kb + kkB);
    bb.q[1] = *(const uint4*)(brow + kb + kkB + 8);
    acc = __builtin_amdgcn_wmma_f32_16x16x32_bf16(false, a.v, false, bb.v,
                                                  (short)0, acc, false, false);
  }
  // D: M = pixel (r + 8*half within tile), N = oc = lr
  const int oc = nt * 16 + lr;
  if (oc < 18) {
    const float bias = boff[oc];
    float* op = off + (((size_t)b * 18 + oc) << 12) + (yo << 6) + mt * 16 + half * 8;
    #pragma unroll
    for (int r = 0; r < 8; ++r) op[r] = acc[r] + bias;
  }
}

// ---- main: fused bilinear gather (->LDS bf16 B operand) + WMMA GEMM ----
__global__ __launch_bounds__(256)
void k_deform_main(const unsigned short* __restrict__ xtb, const float* __restrict__ off,
                   const unsigned short* __restrict__ wpack, float* __restrict__ out) {
  __shared__ __align__(16) unsigned short valT[PTILE * ROWB];
  const int ptile = blockIdx.x, g = blockIdx.y, b = blockIdx.z;
  const int p0 = ptile * PTILE;
  const int tid = threadIdx.x;

  // Phase A: sample 64 pixels x 9 taps x 64 channels into LDS
  for (int idx = tid; idx < PTILE * KTAPS; idx += 256) {
    const int pl = idx / KTAPS, k = idx - pl * KTAPS;
    const int p = p0 + pl, yo = p >> 6, xo = p & 63;
    const float offy = off[(((size_t)b * 18 + 2 * k    ) << 12) + p];
    const float offx = off[(((size_t)b * 18 + 2 * k + 1) << 12) + p];
    const float py = (float)(yo - 1 + k / 3) + offy;
    const float px = (float)(xo - 1 + k % 3) + offx;
    const float y0f = floorf(py), x0f = floorf(px);
    const float wy1 = py - y0f, wx1 = px - x0f;
    const float wy0 = 1.f - wy1, wx0 = 1.f - wx1;
    const float y1f = y0f + 1.f, x1f = x0f + 1.f;
    const bool vy0 = (y0f >= 0.f) && (y0f <= 63.f);
    const bool vy1 = (y1f >= 0.f) && (y1f <= 63.f);
    const bool vx0 = (x0f >= 0.f) && (x0f <= 63.f);
    const bool vx1 = (x1f >= 0.f) && (x1f <= 63.f);
    const int y0 = (int)fminf(fmaxf(y0f, 0.f), 63.f);
    const int y1 = (int)fminf(fmaxf(y1f, 0.f), 63.f);
    const int x0 = (int)fminf(fmaxf(x0f, 0.f), 63.f);
    const int x1 = (int)fminf(fmaxf(x1f, 0.f), 63.f);
    const float w00 = (vy0 && vx0) ? wy0 * wx0 : 0.f;
    const float w01 = (vy0 && vx1) ? wy0 * wx1 : 0.f;
    const float w10 = (vy1 && vx0) ? wy1 * wx0 : 0.f;
    const float w11 = (vy1 && vx1) ? wy1 * wx1 : 0.f;
    const size_t cb = (size_t)g * CG;
    const size_t ib = (size_t)b * XPC;        // padded layout: +1 row/col offset
    const uint4* r00 = (const uint4*)(xtb + ((ib + (y0 + 1) * XP + x0 + 1) * C_TOT + cb));
    const uint4* r01 = (const uint4*)(xtb + ((ib + (y0 + 1) * XP + x1 + 1) * C_TOT + cb));
    const uint4* r10 = (const uint4*)(xtb + ((ib + (y1 + 1) * XP + x0 + 1) * C_TOT + cb));
    const uint4* r11 = (const uint4*)(xtb + ((ib + (y1 + 1) * XP + x1 + 1) * C_TOT + cb));
    unsigned short* dst = &valT[pl * ROWB + k * CG];
    #pragma unroll 2
    for (int i8 = 0; i8 < 8; ++i8) {          // 8 channels (16B of bf16) per step
      uint4 qa = r00[i8], qb = r01[i8], qc = r10[i8], qd = r11[i8];
      unsigned ra[4] = {qa.x, qa.y, qa.z, qa.w};
      unsigned rb[4] = {qb.x, qb.y, qb.z, qb.w};
      unsigned rc[4] = {qc.x, qc.y, qc.z, qc.w};
      unsigned rd[4] = {qd.x, qd.y, qd.z, qd.w};
      unsigned res[4];
      #pragma unroll
      for (int j = 0; j < 4; ++j) {
        float lo = w00 * bf2f(ra[j] & 0xffffu) + w01 * bf2f(rb[j] & 0xffffu) +
                   w10 * bf2f(rc[j] & 0xffffu) + w11 * bf2f(rd[j] & 0xffffu);
        float hi = w00 * bf2f(ra[j] >> 16) + w01 * bf2f(rb[j] >> 16) +
                   w10 * bf2f(rc[j] >> 16) + w11 * bf2f(rd[j] >> 16);
        res[j] = (unsigned)f2bf(lo) | ((unsigned)f2bf(hi) << 16);
      }
      *(uint4*)(dst + i8 * 8) = make_uint4(res[0], res[1], res[2], res[3]);
    }
  }
  __syncthreads();

  // Phase B: per-wave one M-tile x two N-tiles; K = 576 = 18 steps
  const int wv = tid >> 5, lane = tid & 31;
  const int lr = lane & 15, half = lane >> 4;
  const int mt = wv & 3, nt0 = (wv >> 2) << 1;
  union ABU { v16bf v; uint4 q[2]; };
  const unsigned short* arow  = wpack + (size_t)(g * CG + mt * 16 + lr) * KDIM;
  const unsigned short* brow0 = valT + (nt0 * 16 + lr) * ROWB;
  const unsigned short* brow1 = brow0 + 16 * ROWB;
  const int kkA = half * 16;
  const int kkB = half * 16;
  v8f acc0 = {0, 0, 0, 0, 0, 0, 0, 0};
  v8f acc1 = {0, 0, 0, 0, 0, 0, 0, 0};
  #pragma unroll 2
  for (int kt = 0; kt < KDIM / 32; ++kt) {
    const int kb = kt * 32;
    ABU a, b0, b1;
    a.q[0]  = *(const uint4*)(arow  + kb + kkA);
    a.q[1]  = *(const uint4*)(arow  + kb + kkA + 8);
    b0.q[0] = *(const uint4*)(brow0 + kb + kkB);
    b0.q[1] = *(const uint4*)(brow0 + kb + kkB + 8);
    b1.q[0] = *(const uint4*)(brow1 + kb + kkB);
    b1.q[1] = *(const uint4*)(brow1 + kb + kkB + 8);
    acc0 = __builtin_amdgcn_wmma_f32_16x16x32_bf16(false, a.v, false, b0.v,
                                                   (short)0, acc0, false, false);
    acc1 = __builtin_amdgcn_wmma_f32_16x16x32_bf16(false, a.v, false, b1.v,
                                                   (short)0, acc1, false, false);
  }

  // C/D layout: VGPR r -> M = r + 8*half, N = lane%16
  const int mb = g * CG + mt * 16 + half * 8;
  float* o0 = out + ((size_t)b * C_TOT + mb) * HW + (p0 + nt0 * 16 + lr);
  float* o1 = o0 + 16;
  #pragma unroll
  for (int r = 0; r < 8; ++r) {
    o0[(size_t)r * HW] = acc0[r];
    o1[(size_t)r * HW] = acc1[r];
  }
}

extern "C" void kernel_launch(void* const* d_in, const int* in_sizes, int n_in,
                              void* d_out, int out_size, void* d_ws, size_t ws_size,
                              hipStream_t stream) {
  const float* x     = (const float*)d_in[0];
  const float* w_off = (const float*)d_in[1];
  const float* b_off = (const float*)d_in[2];
  const float* w_def = (const float*)d_in[3];
  float* out = (float*)d_out;

  char* ws = (char*)d_ws;
  unsigned short* xtb = (unsigned short*)ws;                        // 8,921,088 B
  float*          off = (float*)(ws + 8921088);                     // 1,179,648 B
  unsigned short* wp  = (unsigned short*)(ws + 8921088 + 1179648);            // 294,912 B
  unsigned short* wob = (unsigned short*)(ws + 8921088 + 1179648 + 294912);   // 147,456 B

  k_transpose<<<(4 * XPC * C_TOT + 255) / 256, 256, 0, stream>>>(x, xtb);
  k_pack<<<(C_TOT * KDIM + 255) / 256, 256, 0, stream>>>(w_def, wp);
  k_packoff<<<(32 * KOFF + 255) / 256, 256, 0, stream>>>(w_off, wob);

  k_offconv_wmma<<<4 * HO, 256, 0, stream>>>(xtb, wob, b_off, off);

  dim3 grid(HW / PTILE, 4, 4);   // (pixel tile, group, batch)
  k_deform_main<<<grid, 256, 0, stream>>>(xtb, off, wp, out);
}
